// DualCompressor_52785148068385
// MI455X (gfx1250) — matrix operand order repaired and checked by reading, not compiled, not run on hardware
//
#include <hip/hip_runtime.h>
#include <math.h>

typedef __attribute__((ext_vector_type(16))) _Float16 v16h;
typedef __attribute__((ext_vector_type(8)))  float    v8f;
typedef __attribute__((ext_vector_type(4)))  float    f32x4;

#define V_TOTAL          100000
#define KDIM             64
#define MH               16
#define TILES_PER_TENSOR (V_TOTAL / 16)          /* 6250, exact */
#define TOTAL_TILES      (2 * TILES_PER_TENSOR)  /* 12500 */

// y[v,m] = rowmax[v] + log( sum_k exp(phi[v,k]-rowmax[v]) * exp(alpha[m,k]) )
// The inner sum is a 16x64 @ 64x16 GEMM done with two v_wmma_f32_16x16x32_f16.
__global__ __launch_bounds__(256) void dual_compress_wmma(
    const float* __restrict__ d_out_t,   // (V,64) input for fwd
    const float* __restrict__ d_in_t,    // (V,64) input for bwd
    const float* __restrict__ alpha_fwd, // (16,64)
    const float* __restrict__ alpha_bwd, // (16,64)
    float* __restrict__ out)             // concat: y_fwd (V,16) then y_bwd (V,16)
{
    const int lane   = threadIdx.x & 31;
    const int waveId = blockIdx.x * (blockDim.x >> 5) + (threadIdx.x >> 5);
    if (waveId >= TOTAL_TILES) return;   // wave-uniform: EXEC stays all-ones below

    const int n    = lane & 15;          // B column / output N / alpha row m
    const int half = lane >> 4;          // which 16-lane group

    const int tensor = (waveId >= TILES_PER_TENSOR) ? 1 : 0;
    const int tile   = tensor ? (waveId - TILES_PER_TENSOR) : waveId;
    const float* __restrict__ phi   = tensor ? d_in_t    : d_out_t;
    const float* __restrict__ alpha = tensor ? alpha_bwd : alpha_fwd;

    // ---- B = exp(alpha)^T in WMMA B layout (32x16 f16 per K-chunk).
    // Lane L, element e of chunk c  <->  B[k = c*32 + 8*half + e + (e & 8)][n].
    v16h b0, b1;
#pragma unroll
    for (int e = 0; e < 16; ++e) {
        b0[e] = (_Float16)__expf(alpha[n * KDIM + (     8 * half + e + (e & 8))]);
        b1[e] = (_Float16)__expf(alpha[n * KDIM + (32 + 8 * half + e + (e & 8))]);
    }

    // ---- Load this wave's 16x64 phi tile in WMMA A layout.
    // Lane L holds row = L%16; chunk c covers K = c*32 + {8h..8h+7, 16+8h..16+8h+7}.
    const size_t rowBase = (size_t)(tile * 16 + n) * KDIM;
    float vals[32];
    float lmax = -INFINITY;
#pragma unroll
    for (int c = 0; c < 2; ++c) {
        const int k0 = c * 32 + 8 * half;        // 32B-aligned
        const int k1 = c * 32 + 16 + 8 * half;   // 16B-aligned
        f32x4 x0 = *(const f32x4*)(phi + rowBase + k0);
        f32x4 x1 = *(const f32x4*)(phi + rowBase + k0 + 4);
        f32x4 x2 = *(const f32x4*)(phi + rowBase + k1);
        f32x4 x3 = *(const f32x4*)(phi + rowBase + k1 + 4);
#pragma unroll
        for (int j = 0; j < 4; ++j) {
            vals[c * 16 + 0  + j] = x0[j];
            vals[c * 16 + 4  + j] = x1[j];
            vals[c * 16 + 8  + j] = x2[j];
            vals[c * 16 + 12 + j] = x3[j];
        }
#pragma unroll
        for (int j = 0; j < 16; ++j) lmax = fmaxf(lmax, vals[c * 16 + j]);
    }
    // Full row max: lanes L and L^16 jointly hold one row.
    const float rmax = fmaxf(lmax, __shfl_xor(lmax, 16, 32));

    // ---- A = exp(phi - rowmax) packed to f16.
    v16h a0, a1;
#pragma unroll
    for (int e = 0; e < 16; ++e) {
        a0[e] = (_Float16)__expf(vals[e]      - rmax);
        a1[e] = (_Float16)__expf(vals[16 + e] - rmax);
    }

    // ---- Two K=32 WMMAs accumulate the full K=64 dot products in f32.
    v8f acc = {};
    acc = __builtin_amdgcn_wmma_f32_16x16x32_f16(false, a0, false, b0,
                                                 (short)0, acc, false, false);
    acc = __builtin_amdgcn_wmma_f32_16x16x32_f16(false, a1, false, b1,
                                                 (short)0, acc, false, false);

    // ---- y = rowmax + log(acc); D layout: VGPR r on lane L = D[r+8*half][n].
    float* __restrict__ outT =
        out + (size_t)tensor * V_TOTAL * MH + (size_t)tile * 16 * MH;
#pragma unroll
    for (int r = 0; r < 8; ++r) {
        const int M = r + 8 * half;
        const float rm = __shfl(rmax, M, 32);   // row M's max lives in lane M
        outT[M * MH + n] = rm + __logf(acc[r]);
    }
}

extern "C" void kernel_launch(void* const* d_in, const int* in_sizes, int n_in,
                              void* d_out, int out_size, void* d_ws, size_t ws_size,
                              hipStream_t stream) {
    const float* d_out_t = (const float*)d_in[0];
    const float* d_in_t  = (const float*)d_in[1];
    const float* a_fwd   = (const float*)d_in[2];
    const float* a_bwd   = (const float*)d_in[3];
    float* out = (float*)d_out;

    const int wavesPerBlock = 8;                       // 256 threads, wave32
    const int blocks = (TOTAL_TILES + wavesPerBlock - 1) / wavesPerBlock; // 1563
    dual_compress_wmma<<<blocks, 256, 0, stream>>>(d_out_t, d_in_t, a_fwd, a_bwd, out);
}
